// LinearAttention_19628000543020
// MI455X (gfx1250) — matrix operand order repaired and checked by reading, not compiled, not run on hardware
//
#include <hip/hip_runtime.h>
#include <hip/hip_bf16.h>

// ---------------------------------------------------------------------------
// Linear attention on MI455X (gfx1250): WMMA f16 (fp32 accum) everywhere.
//   B=4, L=8192, E=512, H=8, D=64, BL=32768
// ---------------------------------------------------------------------------

typedef __attribute__((ext_vector_type(16))) _Float16 v16h;
typedef __attribute__((ext_vector_type(8)))  float    v8f;

#define B_  4
#define L_  8192
#define E_  512
#define H_  8
#define D_  64
#define BL_ (B_ * L_)
#define KC_ 32    // WMMA K chunk (f16)
#define PM_ 128   // proj kernel M tile
#define TM_ 64    // kv/combine M tile
#define TS_ 72    // f16 transposed-tile stride (144B = 9*16B, keeps b128 align)

__device__ __forceinline__ v8f wmma_f16(v16h a, v16h b, v8f c) {
    return __builtin_amdgcn_wmma_f32_16x16x32_f16(false, a, false, b,
                                                  (short)0, c, false, false);
}

// A fragment (16x32 f16) from row-major f16 LDS tile, K-chunk base kc
template <int STRIDE>
__device__ __forceinline__ v16h load_a_frag(const _Float16 (*tile)[STRIDE],
                                            int m0, int kc, int lane) {
    v16h a;
    const int r  = m0 + (lane & 15);
    const int kb = kc + (lane >> 4) * 16;
#pragma unroll
    for (int i = 0; i < 16; ++i) a[i] = tile[r][kb + i];
    return a;
}

// B fragment (32x16 f16) from "N-major" f16 LDS tile (tile[n][k] = B[k][n])
template <int STRIDE>
__device__ __forceinline__ v16h load_b_frag(const _Float16 (*tile)[STRIDE],
                                            int n0, int kc, int lane) {
    v16h b;
    const int c  = n0 + (lane & 15);
    const int kb = kc + (lane >> 4) * 16;
#pragma unroll
    for (int i = 0; i < 16; ++i) b[i] = tile[c][kb + i];
    return b;
}

// ---------------------------------------------------------------------------
__global__ void zero_kernel(float* __restrict__ p, int n) {
    int i = blockIdx.x * blockDim.x + threadIdx.x;
    if (i < n) p[i] = 0.0f;
}

// ---------------------------------------------------------------------------
// Y[row, head*64 + n] = (X @ W^T + bias), optional per-head row softmax.
// grid = (BL/128, H), block = 256 (8 waves); wave: 2 M-tiles x 2 N-tiles.
// ---------------------------------------------------------------------------
__global__ __launch_bounds__(256) void proj_kernel(
    const float* __restrict__ X, const float* __restrict__ W,
    const float* __restrict__ bias, float* __restrict__ Y, int doSoftmax) {
    __shared__ _Float16 Xs[PM_][KC_ + 8];
    __shared__ _Float16 Ws[D_][KC_ + 8];
    __shared__ float    St[PM_][D_ + 1];

    const int row0 = blockIdx.x * PM_;
    const int col0 = blockIdx.y * D_;
    const int tid  = threadIdx.x;
    const int wid  = tid >> 5;
    const int lane = tid & 31;
    const int m0   = (wid >> 1) * 32;        // wave owns m0..m0+31
    const int np   = (wid & 1) * 2;          // N-tile pair (0 or 2)

    v8f acc00 = {}, acc01 = {}, acc10 = {}, acc11 = {};

    const int xr = tid >> 1;                 // X loader: row 0..127
    const int xk = (tid & 1) * 16;           // X loader: k 0/16
    const int wr = tid >> 2;                 // W loader: row 0..63
    const int wk = (tid & 3) * 8;

    for (int kc = 0; kc < E_; kc += KC_) {
        const float* xs = X + (size_t)(row0 + xr) * E_ + kc + xk;
        const float* ws = W + (size_t)(col0 + wr) * E_ + kc + wk;
        if (kc + KC_ < E_) { __builtin_prefetch(xs + KC_, 0, 0);
                             __builtin_prefetch(ws + KC_, 0, 0); }
#pragma unroll
        for (int i = 0; i < 16; ++i) Xs[xr][xk + i] = (_Float16)xs[i];
#pragma unroll
        for (int i = 0; i < 8; ++i)  Ws[wr][wk + i] = (_Float16)ws[i];
        __syncthreads();

        v16h a0 = load_a_frag<KC_ + 8>(Xs, m0,      0, lane);
        v16h a1 = load_a_frag<KC_ + 8>(Xs, m0 + 16, 0, lane);
        v16h b0 = load_b_frag<KC_ + 8>(Ws, (np + 0) * 16, 0, lane);
        v16h b1 = load_b_frag<KC_ + 8>(Ws, (np + 1) * 16, 0, lane);
        acc00 = wmma_f16(a0, b0, acc00);
        acc01 = wmma_f16(a0, b1, acc01);
        acc10 = wmma_f16(a1, b0, acc10);
        acc11 = wmma_f16(a1, b1, acc11);
        __syncthreads();
    }

    // stage results + bias into LDS
    {
        const int n0 = (np + 0) * 16 + (lane & 15);
        const int n1 = (np + 1) * 16 + (lane & 15);
        const float bb0 = bias[col0 + n0];
        const float bb1 = bias[col0 + n1];
        const int mb = m0 + (lane >> 4) * 8;
#pragma unroll
        for (int i = 0; i < 8; ++i) {
            St[mb + i][n0]      = acc00[i] + bb0;
            St[mb + i][n1]      = acc01[i] + bb1;
            St[mb + 16 + i][n0] = acc10[i] + bb0;
            St[mb + 16 + i][n1] = acc11[i] + bb1;
        }
    }
    __syncthreads();

    if (doSoftmax) {
        if (tid < PM_) {
            float mx = -3.4e38f;
#pragma unroll
            for (int n = 0; n < D_; ++n) mx = fmaxf(mx, St[tid][n]);
            float s = 0.0f;
#pragma unroll
            for (int n = 0; n < D_; ++n) {
                float e = __expf(St[tid][n] - mx);
                St[tid][n] = e; s += e;
            }
            const float inv = 1.0f / s;
#pragma unroll
            for (int n = 0; n < D_; ++n) St[tid][n] *= inv;
        }
        __syncthreads();
    }

    {
        const int r  = tid >> 1;
        const int c0 = (tid & 1) * 32;
        float* dst = Y + (size_t)(row0 + r) * E_ + col0 + c0;
#pragma unroll
        for (int i = 0; i < 32; ++i) dst[i] = St[r][c0 + i];
    }
}

// ---------------------------------------------------------------------------
// Fused: k = softmax(F@Wk^T+bk), v = F@Wv^T+bv,
//   kv[b,h] += k^T v (64x64), ksum[b,h] += colsum(k).
// grid = (BL/64, H), block = 256
// ---------------------------------------------------------------------------
__global__ __launch_bounds__(256) void kv_kernel(
    const float* __restrict__ F, const float* __restrict__ Wk,
    const float* __restrict__ bk, const float* __restrict__ Wv,
    const float* __restrict__ bv, float* __restrict__ KV,
    float* __restrict__ KSUM) {
    __shared__ _Float16 Xs[TM_][KC_ + 8];
    __shared__ _Float16 Wks[D_][KC_ + 8];
    __shared__ _Float16 Wvs[D_][KC_ + 8];
    __shared__ float    Ks[TM_][D_ + 1];     // k rows (f32) for softmax/ksum
    __shared__ _Float16 KsT[D_][TS_];        // k^T as f16 (A operand)
    __shared__ _Float16 VsT[D_][TS_];        // v n-major f16 (B operand)

    const int row0 = blockIdx.x * TM_;
    const int head = blockIdx.y;
    const int col0 = head * D_;
    const int b    = row0 / L_;
    const int bh   = b * H_ + head;
    const int tid  = threadIdx.x;
    const int wid  = tid >> 5;
    const int lane = tid & 31;
    const int mt   = (wid >> 1) * 16;
    const int np   = (wid & 1) * 2;

    v8f ak0 = {}, ak1 = {}, av0 = {}, av1 = {};

    const int lr = tid >> 2;
    const int lk = (tid & 3) * 8;

    // ---- phase 1: k-lin and v-lin projections (WMMA over K=512) ----
    for (int kc = 0; kc < E_; kc += KC_) {
        const float* xs = F  + (size_t)(row0 + lr) * E_ + kc + lk;
        const float* wk = Wk + (size_t)(col0 + lr) * E_ + kc + lk;
        const float* wv = Wv + (size_t)(col0 + lr) * E_ + kc + lk;
        if (kc + KC_ < E_) __builtin_prefetch(xs + KC_, 0, 0);
#pragma unroll
        for (int i = 0; i < 8; ++i) {
            Xs[lr][lk + i]  = (_Float16)xs[i];
            Wks[lr][lk + i] = (_Float16)wk[i];
            Wvs[lr][lk + i] = (_Float16)wv[i];
        }
        __syncthreads();

        v16h a    = load_a_frag<KC_ + 8>(Xs, mt, 0, lane);
        v16h bk0f = load_b_frag<KC_ + 8>(Wks, (np + 0) * 16, 0, lane);
        v16h bk1f = load_b_frag<KC_ + 8>(Wks, (np + 1) * 16, 0, lane);
        v16h bv0f = load_b_frag<KC_ + 8>(Wvs, (np + 0) * 16, 0, lane);
        v16h bv1f = load_b_frag<KC_ + 8>(Wvs, (np + 1) * 16, 0, lane);
        ak0 = wmma_f16(a, bk0f, ak0);
        ak1 = wmma_f16(a, bk1f, ak1);
        av0 = wmma_f16(a, bv0f, av0);
        av1 = wmma_f16(a, bv1f, av1);
        __syncthreads();
    }

    // stage: k rows (f32, for softmax) and v transposed (f16 B operand)
    {
        const int n0 = (np + 0) * 16 + (lane & 15);
        const int n1 = (np + 1) * 16 + (lane & 15);
        const float bk0b = bk[col0 + n0], bk1b = bk[col0 + n1];
        const float bv0b = bv[col0 + n0], bv1b = bv[col0 + n1];
        const int mb = mt + (lane >> 4) * 8;
#pragma unroll
        for (int i = 0; i < 8; ++i) {
            Ks[mb + i][n0]  = ak0[i] + bk0b;
            Ks[mb + i][n1]  = ak1[i] + bk1b;
            VsT[n0][mb + i] = (_Float16)(av0[i] + bv0b);
            VsT[n1][mb + i] = (_Float16)(av1[i] + bv1b);
        }
    }
    __syncthreads();

    // row softmax of k (f32)
    if (tid < TM_) {
        float mx = -3.4e38f;
#pragma unroll
        for (int n = 0; n < D_; ++n) mx = fmaxf(mx, Ks[tid][n]);
        float s = 0.0f;
#pragma unroll
        for (int n = 0; n < D_; ++n) {
            float e = __expf(Ks[tid][n] - mx);
            Ks[tid][n] = e; s += e;
        }
        const float inv = 1.0f / s;
#pragma unroll
        for (int n = 0; n < D_; ++n) Ks[tid][n] *= inv;
    }
    __syncthreads();

    // ksum partial (f32, column sums)
    if (tid < D_) {
        float s = 0.0f;
#pragma unroll
        for (int r = 0; r < TM_; ++r) s += Ks[r][tid];
        atomicAdd(&KSUM[(size_t)bh * D_ + tid], s);
    }

    // transpose-convert Ks -> KsT (f16): contiguous vectorizable writes
    {
        const int d  = tid >> 2;
        const int r0 = (tid & 3) * 16;
#pragma unroll
        for (int i = 0; i < 16; ++i) KsT[d][r0 + i] = (_Float16)Ks[r0 + i][d];
    }
    __syncthreads();

    // ---- phase 2: kv partial = KsT (A) x VsT (B), K over 64 rows ----
    v8f p0 = {}, p1 = {};
#pragma unroll
    for (int kc = 0; kc < TM_; kc += KC_) {
        v16h a   = load_a_frag<TS_>(KsT, mt, kc, lane);
        v16h b0f = load_b_frag<TS_>(VsT, (np + 0) * 16, kc, lane);
        v16h b1f = load_b_frag<TS_>(VsT, (np + 1) * 16, kc, lane);
        p0 = wmma_f16(a, b0f, p0);
        p1 = wmma_f16(a, b1f, p1);
    }

    // accumulate kv into global
    {
        const int n0 = (np + 0) * 16 + (lane & 15);
        const int n1 = (np + 1) * 16 + (lane & 15);
        const int mb = mt + (lane >> 4) * 8;
        float* kvb = KV + (size_t)bh * D_ * D_;
#pragma unroll
        for (int i = 0; i < 8; ++i) {
            atomicAdd(&kvb[(mb + i) * D_ + n0], p0[i]);
            atomicAdd(&kvb[(mb + i) * D_ + n1], p1[i]);
        }
    }
}

// ---------------------------------------------------------------------------
// res = q + 0.5*(alpha0*(q@kv0) + alpha1*(q@kv1)), in-place over q.
// grid = (BL/64, H), block = 256
// ---------------------------------------------------------------------------
__global__ __launch_bounds__(256) void combine_kernel(
    float* __restrict__ Q, const float* __restrict__ KV0,
    const float* __restrict__ KS0, const float* __restrict__ KV1,
    const float* __restrict__ KS1) {
    __shared__ float    Qs[TM_][D_ + 1];     // f32 (alpha + residual)
    __shared__ _Float16 Qh[TM_][TS_];        // f16 row-major (A operand)
    __shared__ _Float16 M0[D_][TS_];         // kv0 n-major f16 (B operand)
    __shared__ _Float16 M1[D_][TS_];         // kv1 n-major f16 (B operand)
    __shared__ float    Al0[TM_];
    __shared__ float    Al1[TM_];

    const int row0 = blockIdx.x * TM_;
    const int head = blockIdx.y;
    const int col0 = head * D_;
    const int b    = row0 / L_;
    const int bh   = b * H_ + head;
    const int tid  = threadIdx.x;
    const int wid  = tid >> 5;
    const int lane = tid & 31;
    const int mt   = (wid >> 1) * 16;
    const int np   = (wid & 1) * 2;

    // load q tile (f32 + f16 copies); kv matrices transposed into f16
    {
        const int r  = tid >> 2;
        const int nq = (tid & 3) * 16;
        const float* qsrc = Q + (size_t)(row0 + r) * E_ + col0 + nq;
#pragma unroll
        for (int i = 0; i < 16; ++i) {
            const float qv = qsrc[i];
            Qs[r][nq + i] = qv;
            Qh[r][nq + i] = (_Float16)qv;
        }
        // M*T[n][k] = KV[k*D + n]: strided global reads, contiguous LDS writes
        const int n  = tid >> 2;
        const int k0 = (tid & 3) * 16;
        const float* kv0p = KV0 + (size_t)bh * D_ * D_ + n;
        const float* kv1p = KV1 + (size_t)bh * D_ * D_ + n;
#pragma unroll
        for (int i = 0; i < 16; ++i) {
            M0[n][k0 + i] = (_Float16)kv0p[(k0 + i) * D_];
            M1[n][k0 + i] = (_Float16)kv1p[(k0 + i) * D_];
        }
    }
    __syncthreads();

    // alpha per row
    if (tid < TM_) {
        const float* s0 = KS0 + (size_t)bh * D_;
        const float* s1 = KS1 + (size_t)bh * D_;
        float d0 = 0.0f, d1 = 0.0f;
#pragma unroll
        for (int n = 0; n < D_; ++n) {
            const float qv = Qs[tid][n];
            d0 += qv * s0[n];
            d1 += qv * s1[n];
        }
        Al0[tid] = 1.0f / d0;
        Al1[tid] = 1.0f / d1;
    }
    __syncthreads();

    // y0 = q@kv0, y1 = q@kv1 via WMMA (K = 64)
    v8f y00 = {}, y01 = {}, y10 = {}, y11 = {};
#pragma unroll
    for (int kc = 0; kc < D_; kc += KC_) {
        v16h a   = load_a_frag<TS_>(Qh, mt, kc, lane);
        v16h b00 = load_b_frag<TS_>(M0, (np + 0) * 16, kc, lane);
        v16h b01 = load_b_frag<TS_>(M0, (np + 1) * 16, kc, lane);
        v16h b10 = load_b_frag<TS_>(M1, (np + 0) * 16, kc, lane);
        v16h b11 = load_b_frag<TS_>(M1, (np + 1) * 16, kc, lane);
        y00 = wmma_f16(a, b00, y00);
        y01 = wmma_f16(a, b01, y01);
        y10 = wmma_f16(a, b10, y10);
        y11 = wmma_f16(a, b11, y11);
    }

    // res = q + 0.5*(a0*y0 + a1*y1) -> write back over Q
    {
        const int n0 = (np + 0) * 16 + (lane & 15);
        const int n1 = (np + 1) * 16 + (lane & 15);
        const int mb = mt + (lane >> 4) * 8;
#pragma unroll
        for (int i = 0; i < 8; ++i) {
            const int m  = mb + i;
            const float a0 = Al0[m], a1 = Al1[m];
            float* dst = Q + (size_t)(row0 + m) * E_ + col0;
            dst[n0] = Qs[m][n0] + 0.5f * (a0 * y00[i] + a1 * y10[i]);
            dst[n1] = Qs[m][n1] + 0.5f * (a0 * y01[i] + a1 * y11[i]);
        }
    }
}

// ---------------------------------------------------------------------------
extern "C" void kernel_launch(void* const* d_in, const int* in_sizes, int n_in,
                              void* d_out, int out_size, void* d_ws,
                              size_t ws_size, hipStream_t stream) {
    const float* query = (const float*)d_in[0];
    const float* f0    = (const float*)d_in[1];
    const float* f1    = (const float*)d_in[2];
    const float* Wq    = (const float*)d_in[3];
    const float* bq    = (const float*)d_in[4];
    const float* Wk0   = (const float*)d_in[5];
    const float* bk0   = (const float*)d_in[6];
    const float* Wv0   = (const float*)d_in[7];
    const float* bv0   = (const float*)d_in[8];
    const float* Wk1   = (const float*)d_in[9];
    const float* bk1   = (const float*)d_in[10];
    const float* Wv1   = (const float*)d_in[11];
    const float* bv1   = (const float*)d_in[12];
    const float* Wo    = (const float*)d_in[13];
    const float* bo    = (const float*)d_in[14];
    float* out = (float*)d_out;

    // workspace layout (floats): q/res [BL*E], kv0, ks0, kv1, ks1
    float* q   = (float*)d_ws;
    float* kv0 = q   + (size_t)BL_ * E_;
    float* ks0 = kv0 + (size_t)B_ * H_ * D_ * D_;
    float* kv1 = ks0 + (size_t)B_ * H_ * D_;
    float* ks1 = kv1 + (size_t)B_ * H_ * D_ * D_;

    const int nAccum = 2 * (B_ * H_ * D_ * D_) + 2 * (B_ * H_ * D_);
    const dim3 gproj(BL_ / PM_, H_);
    const dim3 gtile(BL_ / TM_, H_);
    const dim3 blk(256);

    zero_kernel<<<(nAccum + 255) / 256, 256, 0, stream>>>(kv0, nAccum);

    // q = softmax(query @ Wq^T + bq) per head
    proj_kernel<<<gproj, blk, 0, stream>>>(query, Wq, bq, q, 1);

    // kv / ksum accumulation for both functions (fused proj+softmax+k^T v)
    kv_kernel<<<gtile, blk, 0, stream>>>(f0, Wk0, bk0, Wv0, bv0, kv0, ks0);
    kv_kernel<<<gtile, blk, 0, stream>>>(f1, Wk1, bk1, Wv1, bv1, kv1, ks1);

    // res = q + 0.5*(alpha0*q@kv0 + alpha1*q@kv1)  (in-place over q)
    combine_kernel<<<gtile, blk, 0, stream>>>(q, kv0, ks0, kv1, ks1);

    // out = res @ Wo^T + bo
    proj_kernel<<<gproj, blk, 0, stream>>>(q, Wo, bo, out, 0);
}
